// Barrier_Net_13503377178879
// MI455X (gfx1250) — matrix operand order repaired and checked by reading, not compiled, not run on hardware
//
#include <hip/hip_runtime.h>
#include <hip/hip_bf16.h>

// ---------------------------------------------------------------------------
// Barrier-Net fused inference for gfx1250 (MI455X, wave32, WMMA).
//
// Pipeline per launch (all on `stream`, deterministic):
//   0) zero agg[16384*64] in d_ws
//   1) transpose+convert weights f32 -> f16 (N-major) in d_ws
//   2) edge MLP  4->256->256->64 fused, WMMA f32_16x16x32_f16, atomic segment sum
//   3) agent MLP 64->256->256->2 fused, WMMA, writes d_out
//   4) barrier term, atomicAdd into d_out
// ---------------------------------------------------------------------------

typedef __attribute__((ext_vector_type(16))) _Float16 v16h;
typedef __attribute__((ext_vector_type(8)))  _Float16 v8h;
typedef __attribute__((ext_vector_type(8)))  float    v8f;

#define N_AGENTS 16384
#define N_EDGES  524288
#define EPB      64          // edges per block (edge kernel)
#define APB      64          // agents per block (agent kernel)
#define R_BARRIER (1.2f * 0.15f)

// ---- d_ws layout (bytes) ---------------------------------------------------
#define WS_AGG    0                         // 16384*64 f32   = 4194304 B
#define WS_WP2T   (WS_AGG  + 4194304)       // 256x256 f16    = 131072 B
#define WS_WP3T   (WS_WP2T + 131072)        // 64x256  f16    = 32768 B
#define WS_WR1T   (WS_WP3T + 32768)         // 256x64  f16    = 32768 B
#define WS_WR2T   (WS_WR1T + 32768)         // 256x256 f16    = 131072 B
// total ~4.52 MB

// ---------------------------------------------------------------------------
// Fragment loaders for the WMMA operands; A and B both use contiguous k = 16h + s (h = lane/16).
// ---------------------------------------------------------------------------

// 16-bit A matrix 16x32 (MxK), A staged row-major in LDS with stride `lda`.
// lanes 0-15: M=lane,   elems 0-15 => K=kb..kb+15 (kb=kbase)
// lanes16-31: M=lane-16, same with kb=kbase+16.
__device__ inline v16h load_a_frag(const _Float16* A, int lda, int mtile, int kbase, int lane) {
  int m  = mtile * 16 + (lane & 15);
  int kb = kbase + ((lane & 16) ? 16 : 0);
  const _Float16* p = A + m * lda + kb;
  union { v16h v; v8h h[2]; } u;
  u.h[0] = *(const v8h*)(p);        // K = kb .. kb+7
  u.h[1] = *(const v8h*)(p + 8);    // K = kb+8 .. kb+15
  return u.v;
}

// 16-bit B matrix 32x16 (KxN) from f16 weights stored transposed (N-major,
// row stride = Ktot).  lanes 0-15: N=lane, halves = K kbase..kbase+15;
// lanes 16-31: N=lane-16, halves = K kbase+16..kbase+31.
// One contiguous 32-byte load per lane.
__device__ inline v16h load_b_frag(const _Float16* Bt, int Ktot, int ntile, int kbase, int lane) {
  int n  = ntile * 16 + (lane & 15);
  int kb = kbase + ((lane & 16) ? 16 : 0);
  return *(const v16h*)(Bt + n * Ktot + kb);
}

// f32 C/D 16x16: VGPR v holds (M = mtile*16 + v + (lane>=16 ? 8 : 0), N = lane&15).
__device__ inline void store_c_lds(_Float16* C, int ldc, int mtile, int ntile,
                                   v8f acc, const float* bias, int lane, bool relu) {
  int n  = ntile * 16 + (lane & 15);
  float b = bias[n];
  int mb = mtile * 16 + ((lane & 16) ? 8 : 0);
#pragma unroll
  for (int v = 0; v < 8; ++v) {
    float x = acc[v] + b;
    if (relu) x = fmaxf(x, 0.0f);
    C[(mb + v) * ldc + n] = (_Float16)x;
  }
}

// One wave computes MT consecutive 16-row M tiles x one 16-col N tile,
// K = KS*32, A from LDS, B from global (f16, N-major).  B fragment is
// amortized across the MT M-tiles each K step.
template <int MT, int KS>
__device__ inline void gemm_tiles(const _Float16* A, int lda,
                                  const _Float16* Bt, int Ktot,
                                  _Float16* C, int ldc, const float* bias,
                                  int mt0, int nt, int lane, bool relu) {
  v8f acc[MT];
  v8f zero = {};
#pragma unroll
  for (int i = 0; i < MT; ++i) acc[i] = zero;
#pragma unroll
  for (int ks = 0; ks < KS; ++ks) {
    v16h b = load_b_frag(Bt, Ktot, nt, ks * 32, lane);
#pragma unroll
    for (int i = 0; i < MT; ++i) {
      v16h a = load_a_frag(A, lda, mt0 + i, ks * 32, lane);
      acc[i] = __builtin_amdgcn_wmma_f32_16x16x32_f16(
          /*neg_a=*/false, a, /*neg_b=*/false, b,
          /*c_mod=*/(short)0, acc[i], /*reuse_a=*/false, /*reuse_b=*/false);
    }
  }
#pragma unroll
  for (int i = 0; i < MT; ++i) store_c_lds(C, ldc, mt0 + i, nt, acc[i], bias, lane, relu);
}

// ---------------------------------------------------------------------------
// Prep kernels
// ---------------------------------------------------------------------------
__global__ void zero_f32(float* p, int n) {
  int i = blockIdx.x * blockDim.x + threadIdx.x;
  if (i < n) p[i] = 0.0f;
}

// dst[n*K + k] = (f16) src[k*N + n]   (src is row-major KxN)
__global__ void transpose_to_f16(const float* __restrict__ src,
                                 _Float16* __restrict__ dst, int K, int N) {
  int idx = blockIdx.x * blockDim.x + threadIdx.x;
  if (idx < K * N) {
    int k = idx / N, n = idx % N;
    dst[n * K + k] = (_Float16)src[idx];
  }
}

// ---------------------------------------------------------------------------
// Edge MLP: 4 -> 256 -> 256 -> 64, fused + atomic segment-sum into agg.
// 64 edges / block, 256 threads (8 waves).
// ---------------------------------------------------------------------------
__global__ __launch_bounds__(256) void edge_mlp_kernel(
    const float* __restrict__ edge_feats, const int* __restrict__ seg_ids,
    const float* __restrict__ Wp1, const float* __restrict__ bp1,
    const float* __restrict__ bp2, const float* __restrict__ bp3,
    const _Float16* __restrict__ Wp2t, const _Float16* __restrict__ Wp3t,
    float* __restrict__ agg) {
  __shared__ _Float16 h1[EPB * 256];   // 32 KB
  __shared__ _Float16 h2[EPB * 256];   // 32 KB
  __shared__ int      segs[EPB];

  const int tid  = threadIdx.x;
  const int lane = tid & 31;
  const int wave = tid >> 5;
  const int e0   = blockIdx.x * EPB;

  if (tid < EPB) segs[tid] = seg_ids[e0 + tid];

  // ---- layer 1 (K=4): VALU, one output column per thread -------------------
  {
    const int n = tid;  // 0..255
    const float w0 = Wp1[n], w1 = Wp1[256 + n], w2 = Wp1[512 + n], w3 = Wp1[768 + n];
    const float b  = bp1[n];
#pragma unroll 4
    for (int m = 0; m < EPB; ++m) {
      float4 x = ((const float4*)edge_feats)[e0 + m];
      float s = fmaf(x.x, w0, fmaf(x.y, w1, fmaf(x.z, w2, fmaf(x.w, w3, b))));
      h1[m * 256 + n] = (_Float16)fmaxf(s, 0.0f);
    }
  }
  __syncthreads();

  // ---- layer 2: h2 = relu(h1 @ Wp2 + bp2); M=64 N=256 K=256 ----------------
  // wave w owns N tiles {2w, 2w+1}; 4 M tiles each; 8 K steps.
  for (int nt = wave * 2; nt < wave * 2 + 2; ++nt)
    gemm_tiles<4, 8>(h1, 256, Wp2t, 256, h2, 256, bp2, 0, nt, lane, true);
  __syncthreads();

  // ---- layer 3: h3 = h2 @ Wp3 + bp3; M=64 N=64 K=256 -----------------------
  // wave w: nt = w&3, M tiles {2*(w>>2), 2*(w>>2)+1}; epilogue = segment atomics.
  {
    const int nt  = wave & 3;
    const int mt0 = (wave >> 2) * 2;
    v8f acc[2];
    v8f zero = {};
    acc[0] = zero; acc[1] = zero;
#pragma unroll
    for (int ks = 0; ks < 8; ++ks) {
      v16h b = load_b_frag(Wp3t, 256, nt, ks * 32, lane);
#pragma unroll
      for (int i = 0; i < 2; ++i) {
        v16h a = load_a_frag(h2, 256, mt0 + i, ks * 32, lane);
        acc[i] = __builtin_amdgcn_wmma_f32_16x16x32_f16(
            false, a, false, b, (short)0, acc[i], false, false);
      }
    }
    const int n    = nt * 16 + (lane & 15);
    const float bb = bp3[n];
#pragma unroll
    for (int i = 0; i < 2; ++i) {
      const int mb = (mt0 + i) * 16 + ((lane & 16) ? 8 : 0);
#pragma unroll
      for (int v = 0; v < 8; ++v) {
        int s = segs[mb + v];
        atomicAdd(&agg[(size_t)s * 64 + n], acc[i][v] + bb);
      }
    }
  }
}

// ---------------------------------------------------------------------------
// Agent MLP: 64 -> 256 -> 256 -> 2.  64 agents / block, 256 threads (8 waves).
// ---------------------------------------------------------------------------
__global__ __launch_bounds__(256) void agent_mlp_kernel(
    const float* __restrict__ agg,
    const float* __restrict__ br1, const float* __restrict__ br2,
    const float* __restrict__ br3, const float* __restrict__ Wr3,
    const _Float16* __restrict__ Wr1t, const _Float16* __restrict__ Wr2t,
    float* __restrict__ out) {
  __shared__ _Float16 r1[APB * 256];    // 32 KB
  __shared__ _Float16 buf[APB * 256];   // 32 KB: first a0 (64x64), later r2 (64x256)

  const int tid  = threadIdx.x;
  const int lane = tid & 31;
  const int wave = tid >> 5;
  const int a0   = blockIdx.x * APB;

  // stage agg tile as f16 (a0 view: lda = 64)
  for (int idx = tid; idx < APB * 64; idx += 256) {
    int m = idx >> 6, k = idx & 63;
    buf[m * 64 + k] = (_Float16)agg[(size_t)(a0 + m) * 64 + k];
  }
  __syncthreads();

  // r1 = relu(a @ Wr1 + br1); M=64 N=256 K=64 (2 K-steps)
  for (int nt = wave * 2; nt < wave * 2 + 2; ++nt)
    gemm_tiles<4, 2>(buf, 64, Wr1t, 64, r1, 256, br1, 0, nt, lane, true);
  __syncthreads();

  // r2 = relu(r1 @ Wr2 + br2); M=64 N=256 K=256 -> stored back into buf
  for (int nt = wave * 2; nt < wave * 2 + 2; ++nt)
    gemm_tiles<4, 8>(r1, 256, Wr2t, 256, buf, 256, br2, 0, nt, lane, true);
  __syncthreads();

  // out = r2 @ Wr3 + br3; N=2 -> VALU dot products (threads 0..127)
  if (tid < APB * 2) {
    int m = tid >> 1, n = tid & 1;
    float s = br3[n];
    for (int k = 0; k < 256; ++k)
      s = fmaf((float)buf[m * 256 + k], Wr3[k * 2 + n], s);
    out[(size_t)(a0 + m) * 2 + n] = s;
  }
}

// ---------------------------------------------------------------------------
// Barrier term: contrib = -p / (d * (d - 1.2*R_AGENT)); segment-sum into out.
// ---------------------------------------------------------------------------
__global__ __launch_bounds__(256) void barrier_kernel(
    const float* __restrict__ edge_feats, const int* __restrict__ seg_ids,
    float* __restrict__ out) {
  int e = blockIdx.x * blockDim.x + threadIdx.x;
  if (e < N_EDGES) {
    float px = edge_feats[(size_t)e * 4 + 0];
    float py = edge_feats[(size_t)e * 4 + 1];
    float d  = sqrtf(px * px + py * py);
    float f  = -1.0f / (d * (d - R_BARRIER));
    int s = seg_ids[e];
    atomicAdd(&out[(size_t)s * 2 + 0], px * f);
    atomicAdd(&out[(size_t)s * 2 + 1], py * f);
  }
}

// ---------------------------------------------------------------------------
extern "C" void kernel_launch(void* const* d_in, const int* in_sizes, int n_in,
                              void* d_out, int out_size, void* d_ws, size_t ws_size,
                              hipStream_t stream) {
  const float* edge_feats = (const float*)d_in[0];
  const int*   seg_ids    = (const int*)d_in[1];
  const float* Wp1 = (const float*)d_in[2];  const float* bp1 = (const float*)d_in[3];
  const float* Wp2 = (const float*)d_in[4];  const float* bp2 = (const float*)d_in[5];
  const float* Wp3 = (const float*)d_in[6];  const float* bp3 = (const float*)d_in[7];
  const float* Wr1 = (const float*)d_in[8];  const float* br1 = (const float*)d_in[9];
  const float* Wr2 = (const float*)d_in[10]; const float* br2 = (const float*)d_in[11];
  const float* Wr3 = (const float*)d_in[12]; const float* br3 = (const float*)d_in[13];
  float* out = (float*)d_out;

  char* ws = (char*)d_ws;
  float*    agg  = (float*)(ws + WS_AGG);
  _Float16* Wp2t = (_Float16*)(ws + WS_WP2T);
  _Float16* Wp3t = (_Float16*)(ws + WS_WP3T);
  _Float16* Wr1t = (_Float16*)(ws + WS_WR1T);
  _Float16* Wr2t = (_Float16*)(ws + WS_WR2T);

  // 0) zero the aggregation buffer (idempotent per launch)
  zero_f32<<<(N_AGENTS * 64) / 256, 256, 0, stream>>>(agg, N_AGENTS * 64);

  // 1) f32 -> f16 transposed (N-major) weight copies
  transpose_to_f16<<<256, 256, 0, stream>>>(Wp2, Wp2t, 256, 256);
  transpose_to_f16<<<64,  256, 0, stream>>>(Wp3, Wp3t, 256, 64);
  transpose_to_f16<<<64,  256, 0, stream>>>(Wr1, Wr1t, 64,  256);
  transpose_to_f16<<<256, 256, 0, stream>>>(Wr2, Wr2t, 256, 256);

  // 2) fused edge MLP + segment sum
  edge_mlp_kernel<<<N_EDGES / EPB, 256, 0, stream>>>(
      edge_feats, seg_ids, Wp1, bp1, bp2, bp3, Wp2t, Wp3t, agg);

  // 3) fused agent MLP -> d_out
  agent_mlp_kernel<<<N_AGENTS / APB, 256, 0, stream>>>(
      agg, br1, br2, br3, Wr3, Wr1t, Wr2t, out);

  // 4) barrier term accumulated into d_out
  barrier_kernel<<<N_EDGES / 256, 256, 0, stream>>>(edge_feats, seg_ids, out);
}
